// SpatialTransformerModel_26173530702437
// MI455X (gfx1250) — matrix operand order repaired and checked by reading, not compiled, not run on hardware
//
#include <hip/hip_runtime.h>
#include <hip/hip_bf16.h>

// Problem constants
#define B_   8
#define T_   12
#define NKV  325            // nodes
#define KH_  8
#define DH_  64
#define D_   512
#define ROWS (B_*T_*NKV)    // 31200 (= 1950 * 16, exact)
#define NP   352            // keys padded to multiple of 32
#define NT_Q 21             // ceil(325/16) query/key tiles

typedef _Float16 v16h __attribute__((ext_vector_type(16)));
typedef _Float16 v8h  __attribute__((ext_vector_type(8)));
typedef float    v8f  __attribute__((ext_vector_type(8)));

static __device__ inline v8f wmma16(v16h a, v16h b, v8f c) {
  // D = A(16x32 f16) * B(32x16 f16) + C(16x16 f32)
  return __builtin_amdgcn_wmma_f32_16x16x32_f16(false, a, false, b, (short)0, c, false, false);
}

static __device__ inline v16h mk16(v8h lo, v8h hi) {
  v16h r;
#pragma unroll
  for (int i = 0; i < 8; i++) { r[i] = lo[i]; r[i + 8] = hi[i]; }
  return r;
}

// A-matrix 16x32 f16 fragment (ISA 7.12.2): lanes 0-15 hold row M=lane, K {0..7,16..23};
// lanes 16-31 hold row M=lane-16, K {8..15,24..31}. p = row pointer at K-base.
static __device__ inline v16h a_frag(const _Float16* p, int laneHi) {
  v8h lo = *(const v8h*)(p + laneHi * 8);
  v8h hi = *(const v8h*)(p + 16 + laneHi * 8);
  return mk16(lo, hi);
}

// B-matrix 32x16 f16 fragment: lane holds column N=lane%16, K matching A fragment
// (lanes 0-15: K {0..7,16..23}, lanes 16-31: K {8..15,24..31}). p pre-offset to column base + laneHi*16.
static __device__ inline v16h b_frag(const _Float16* p) {
  const int lh = (threadIdx.x >> 4) & 1;
  v8h lo = *(const v8h*)(p - lh * 8);
  v8h hi = *(const v8h*)(p + 16 - lh * 8);
  return mk16(lo, hi);
}

// A fragment built from f32 source with on-the-fly cvt to f16
static __device__ inline v16h a_frag_f32(const float* p, int laneHi) {
  v16h r;
  const float* p0 = p + laneHi * 8;
  const float* p1 = p + 16 + laneHi * 8;
#pragma unroll
  for (int i = 0; i < 8; i++) { r[i] = (_Float16)p0[i]; r[i + 8] = (_Float16)p1[i]; }
  return r;
}

// ---------------------------------------------------------------------------
// Kernel 0: transpose + f32->f16 convert all weights so B-fragments are
// contiguous-K loads. WT layout: [n][k].
// ---------------------------------------------------------------------------
__global__ __launch_bounds__(256) void prep_weights(
    const float* __restrict__ Wq, const float* __restrict__ Wk,
    const float* __restrict__ Wv, const float* __restrict__ Wo1,
    const float* __restrict__ Wo2,
    _Float16* __restrict__ WqT, _Float16* __restrict__ WkT,
    _Float16* __restrict__ WvT, _Float16* __restrict__ Wo1T,
    _Float16* __restrict__ Wo2T) {
  int idx = blockIdx.x * 256 + threadIdx.x;
  if (idx < 1024 * 512) {           // [1024,512] -> [512][1024]
    int k = idx >> 9, n = idx & 511;
    WqT[n * 1024 + k] = (_Float16)Wq[idx];
    WkT[n * 1024 + k] = (_Float16)Wk[idx];
    WvT[n * 1024 + k] = (_Float16)Wv[idx];
  }
  if (idx < 512 * 512) {            // [512,512] -> [512][512]
    int k = idx >> 9, n = idx & 511;
    Wo1T[n * 512 + k] = (_Float16)Wo1[idx];
    Wo2T[n * 512 + k] = (_Float16)Wo2[idx];
  }
}

// ---------------------------------------------------------------------------
// Kernel 1: fused concat([X,STE]) @ W + b, ReLU  (q,k,v projections)
// One wave = one 16x16 output tile; K = 1024 -> 32 WMMA steps.
// q,k stored f16 [row][512]; v stored transposed [bt][kh][d][keyNP] for P@V.
// ---------------------------------------------------------------------------
__global__ __launch_bounds__(256) void proj_kernel(
    const float* __restrict__ Q, const float* __restrict__ Kin,
    const float* __restrict__ V, const float* __restrict__ STE,
    const _Float16* __restrict__ WqT, const _Float16* __restrict__ WkT,
    const _Float16* __restrict__ WvT,
    const float* __restrict__ bq, const float* __restrict__ bk,
    const float* __restrict__ bv,
    _Float16* __restrict__ q16, _Float16* __restrict__ k16,
    _Float16* __restrict__ vT) {
  const int lane = threadIdx.x & 31, laneHi = lane >> 4, ln = lane & 15;
  int tile = blockIdx.x * 8 + (threadIdx.x >> 5);
  int ntile = tile & 31;              // 32 column tiles (512/16)
  int mtile = (tile >> 5) % 1950;     // 1950 row tiles (31200/16)
  int mat   = tile / (32 * 1950);     // 0=q, 1=k, 2=v

  const float* X = (mat == 0) ? Q : (mat == 1) ? Kin : V;
  const _Float16* WT = (mat == 0) ? WqT : (mat == 1) ? WkT : WvT;
  const float* bias = (mat == 0) ? bq : (mat == 1) ? bk : bv;

  const int arow = mtile * 16 + ln;
  const float* xrow = X   + (size_t)arow * D_;
  const float* srow = STE + (size_t)arow * D_;
  const int col = ntile * 16 + ln;
  const _Float16* wrow = WT + (size_t)col * 1024;

  v8f acc = {};
#pragma unroll 4
  for (int kb = 0; kb < 1024; kb += 32) {
    // concat: K-block [kb,kb+32) lies entirely in X (kb<512) or STE (kb>=512)
    const float* ap = (kb < D_) ? (xrow + kb) : (srow + (kb - D_));
    v16h a = a_frag_f32(ap, laneHi);
    v16h b = b_frag(wrow + kb + laneHi * 16);
    acc = wmma16(a, b, acc);
  }

  float bb = bias[col];
  if (mat < 2) {
    _Float16* dst = (mat == 0) ? q16 : k16;
#pragma unroll
    for (int i = 0; i < 8; i++) {
      float v = acc[i] + bb; v = v > 0.f ? v : 0.f;
      int row = mtile * 16 + i + laneHi * 8;
      dst[(size_t)row * D_ + col] = (_Float16)v;
    }
  } else {
    int kh = col >> 6, dd = col & 63;
#pragma unroll
    for (int i = 0; i < 8; i++) {
      float v = acc[i] + bb; v = v > 0.f ? v : 0.f;
      int row = mtile * 16 + i + laneHi * 8;   // row = bt*325 + n
      int bt = row / NKV, n = row % NKV;
      vT[((size_t)bt * D_ + kh * DH_ + dd) * NP + n] = (_Float16)v;
    }
  }
}

// ---------------------------------------------------------------------------
// Kernel 2: attention for one (b,t,head,16-query tile) per wave32 block.
// scores -> LDS, softmax (shfl_xor reductions), probs -> d_out (single pass
// over the 324MB attn output) + f16 LDS pad, then P @ V via WMMA.
// ---------------------------------------------------------------------------
__global__ __launch_bounds__(32) void attn_kernel(
    const _Float16* __restrict__ q16, const _Float16* __restrict__ k16,
    const _Float16* __restrict__ vT, float* __restrict__ attn_out,
    _Float16* __restrict__ x16) {
  __shared__ __align__(16) float     sc[16][NP];   // 22.5 KB
  __shared__ __align__(16) _Float16 p16[16][NP];   // 11.0 KB

  const int lane = threadIdx.x & 31, laneHi = lane >> 4, ln = lane & 15;
  int bid = blockIdx.x;
  int qt = bid % NT_Q; bid /= NT_Q;
  int kh = bid % KH_;  bid /= KH_;
  int t  = bid % T_;   int b = bid / T_;
  const int bt = b * T_ + t;

  // persist Q fragments (d 0..31 and 32..63) in VGPRs
  int nq = qt * 16 + ln;
  int nqc = nq < NKV ? nq : NKV - 1;
  const _Float16* qrp = q16 + ((size_t)(bt * NKV + nqc)) * D_ + kh * DH_;
  v16h qa0 = a_frag(qrp, laneHi);
  v16h qa1 = a_frag(qrp + 32, laneHi);

  // pad scores with large negative so softmax zeros invalid keys
  for (int j = lane; j < 16 * NP; j += 32) (&sc[0][0])[j] = -1.0e30f;

  // scores = (q . k^T) / sqrt(64)
  for (int kt = 0; kt < NT_Q; ++kt) {
    int m = kt * 16 + ln;
    int mc = m < NKV ? m : NKV - 1;
    const _Float16* krp = k16 + ((size_t)(bt * NKV + mc)) * D_ + kh * DH_;
    v16h kb0 = b_frag(krp + laneHi * 16);
    v16h kb1 = b_frag(krp + 32 + laneHi * 16);
    v8f c = {};
    c = wmma16(qa0, kb0, c);
    c = wmma16(qa1, kb1, c);
    if (m < NKV) {
#pragma unroll
      for (int i = 0; i < 8; i++) sc[i + laneHi * 8][m] = c[i] * 0.125f;
    }
  }

  // row softmax + write probs (f32 to d_out, f16 to LDS)
  for (int r = 0; r < 16; r++) {
    float mx = -1.0e30f;
    for (int j = lane; j < NP; j += 32) mx = fmaxf(mx, sc[r][j]);
#pragma unroll
    for (int off = 16; off > 0; off >>= 1) mx = fmaxf(mx, __shfl_xor(mx, off, 32));
    float sum = 0.f;
    for (int j = lane; j < NP; j += 32) {
      float e = __expf(sc[r][j] - mx); sc[r][j] = e; sum += e;
    }
#pragma unroll
    for (int off = 16; off > 0; off >>= 1) sum += __shfl_xor(sum, off, 32);
    float inv = 1.f / sum;
    int nqr = qt * 16 + r;
    float* orow = attn_out +
        ((((size_t)(kh * B_ + b) * T_ + t) * NKV + (nqr < NKV ? nqr : 0)) * NKV);
    for (int j = lane; j < NP; j += 32) {
      float pv = sc[r][j] * inv;
      p16[r][j] = (_Float16)pv;
      if (nqr < NKV && j < NKV) orow[j] = pv;
    }
  }

  // x = P @ V  (B-fragments contiguous thanks to transposed V layout)
  v8f accd[4] = {{}, {}, {}, {}};
  const _Float16* vbase = vT + ((size_t)bt * D_ + kh * DH_) * NP;
  for (int ks = 0; ks < NP; ks += 32) {
    v8h lo = *(const v8h*)&p16[ln][ks + laneHi * 8];
    v8h hi = *(const v8h*)&p16[ln][ks + 16 + laneHi * 8];
    v16h pa = mk16(lo, hi);
#pragma unroll
    for (int dt = 0; dt < 4; dt++) {
      const _Float16* vp = vbase + (size_t)(dt * 16 + ln) * NP + ks + laneHi * 16;
      accd[dt] = wmma16(pa, b_frag(vp), accd[dt]);
    }
  }

#pragma unroll
  for (int dt = 0; dt < 4; dt++) {
#pragma unroll
    for (int i = 0; i < 8; i++) {
      int rq = qt * 16 + i + laneHi * 8;
      if (rq < NKV)
        x16[((size_t)(bt * NKV + rq)) * D_ + kh * DH_ + dt * 16 + ln] =
            (_Float16)accd[dt][i];
    }
  }
}

// ---------------------------------------------------------------------------
// Kernel 3: X = relu(x @ Wo1 + bo1) @ Wo2 + bo2, hidden kept in LDS (f16).
// 16 rows per block, 8 waves each owning 4 column tiles.
// ---------------------------------------------------------------------------
__global__ __launch_bounds__(256) void mlp_kernel(
    const _Float16* __restrict__ x16, const _Float16* __restrict__ Wo1T,
    const _Float16* __restrict__ Wo2T, const float* __restrict__ bo1,
    const float* __restrict__ bo2, float* __restrict__ Xout) {
  __shared__ __align__(16) _Float16 h16[16][D_];   // 16 KB
  const int lane = threadIdx.x & 31, laneHi = lane >> 4, ln = lane & 15;
  const int w = threadIdx.x >> 5;
  const int mtile = blockIdx.x;
  const _Float16* xrow = x16 + (size_t)(mtile * 16 + ln) * D_;

#pragma unroll
  for (int j = 0; j < 4; j++) {
    int col = (w + j * 8) * 16 + ln;
    const _Float16* wrow = Wo1T + (size_t)col * D_;
    v8f acc = {};
#pragma unroll 4
    for (int kb = 0; kb < D_; kb += 32) {
      v16h a = a_frag(xrow + kb, laneHi);
      v16h b = b_frag(wrow + kb + laneHi * 16);
      acc = wmma16(a, b, acc);
    }
    float bb = bo1[col];
#pragma unroll
    for (int i = 0; i < 8; i++) {
      float v = acc[i] + bb; v = v > 0.f ? v : 0.f;
      h16[i + laneHi * 8][col] = (_Float16)v;
    }
  }
  __syncthreads();

#pragma unroll
  for (int j = 0; j < 4; j++) {
    int col = (w + j * 8) * 16 + ln;
    const _Float16* wrow = Wo2T + (size_t)col * D_;
    v8f acc = {};
#pragma unroll 4
    for (int kb = 0; kb < D_; kb += 32) {
      v8h lo = *(const v8h*)&h16[ln][kb + laneHi * 8];
      v8h hi = *(const v8h*)&h16[ln][kb + 16 + laneHi * 8];
      v16h a = mk16(lo, hi);
      v16h b = b_frag(wrow + kb + laneHi * 16);
      acc = wmma16(a, b, acc);
    }
    float bb = bo2[col];
#pragma unroll
    for (int i = 0; i < 8; i++)
      Xout[(size_t)(mtile * 16 + i + laneHi * 8) * D_ + col] = acc[i] + bb;
  }
}

// ---------------------------------------------------------------------------
extern "C" void kernel_launch(void* const* d_in, const int* in_sizes, int n_in,
                              void* d_out, int out_size, void* d_ws, size_t ws_size,
                              hipStream_t stream) {
  (void)in_sizes; (void)n_in; (void)out_size; (void)ws_size;
  const float* Q   = (const float*)d_in[0];
  const float* Kin = (const float*)d_in[1];
  const float* V   = (const float*)d_in[2];
  const float* STE = (const float*)d_in[3];
  // d_in[4] = mask (unused by reference forward)
  const float* Wq  = (const float*)d_in[5];  const float* bq  = (const float*)d_in[6];
  const float* Wk  = (const float*)d_in[7];  const float* bk  = (const float*)d_in[8];
  const float* Wv  = (const float*)d_in[9];  const float* bv  = (const float*)d_in[10];
  const float* Wo1 = (const float*)d_in[11]; const float* bo1 = (const float*)d_in[12];
  const float* Wo2 = (const float*)d_in[13]; const float* bo2 = (const float*)d_in[14];

  float* Xout     = (float*)d_out;                       // [B,T,N,512]
  float* attn_out = Xout + (size_t)ROWS * D_;            // [KH*B,T,N,N]

  char* ws = (char*)d_ws;
  size_t off = 0;
  _Float16* q16 = (_Float16*)(ws + off); off += (size_t)ROWS * D_ * 2;
  _Float16* k16 = (_Float16*)(ws + off); off += (size_t)ROWS * D_ * 2;
  _Float16* x16 = (_Float16*)(ws + off); off += (size_t)ROWS * D_ * 2;
  _Float16* vT  = (_Float16*)(ws + off);
  size_t vT_bytes = (size_t)B_ * T_ * D_ * NP * 2;       off += vT_bytes;
  _Float16* WqT  = (_Float16*)(ws + off); off += (size_t)512 * 1024 * 2;
  _Float16* WkT  = (_Float16*)(ws + off); off += (size_t)512 * 1024 * 2;
  _Float16* WvT  = (_Float16*)(ws + off); off += (size_t)512 * 1024 * 2;
  _Float16* Wo1T = (_Float16*)(ws + off); off += (size_t)512 * 512 * 2;
  _Float16* Wo2T = (_Float16*)(ws + off); off += (size_t)512 * 512 * 2;
  // total ~135 MB of workspace

  // zero transposed-V so padded keys (325..351) contribute nothing to P@V
  hipMemsetAsync(vT, 0, vT_bytes, stream);

  prep_weights<<<2048, 256, 0, stream>>>(Wq, Wk, Wv, Wo1, Wo2,
                                         WqT, WkT, WvT, Wo1T, Wo2T);

  // 3 mats * 1950 row-tiles * 32 col-tiles = 187200 wave-tiles, 8 waves/block
  proj_kernel<<<187200 / 8, 256, 0, stream>>>(Q, Kin, V, STE,
                                              WqT, WkT, WvT, bq, bk, bv,
                                              q16, k16, vT);

  attn_kernel<<<B_ * T_ * KH_ * NT_Q, 32, 0, stream>>>(q16, k16, vT,
                                                       attn_out, x16);

  mlp_kernel<<<1950, 256, 0, stream>>>(x16, Wo1T, Wo2T, bo1, bo2, Xout);
}
